// AttentionLayer_17575006175308
// MI455X (gfx1250) — compile-verified, matrix operand order checked by reading, not run on hardware
//
#include <hip/hip_runtime.h>
#include <hip/hip_bf16.h>

// ---------------------------------------------------------------------------
// MI455X (gfx1250) attention layer: all matmuls on v_wmma_f32_16x16x32_bf16.
// Layout assumptions per CDNA5 ISA 7.12.2 (16-bit A/B, wave32):
//   A 16x32: lane L -> row M=L%16; elems 0..7 = K=(L/16)*8 + j, elems 8..15 =
//            K=16+(L/16)*8 + j  (two contiguous 16B runs along K).
//   B 32x16: mirror of A with N across lanes (column contiguous along K).
//   C/D 16x16 f32: lane L -> N=L%16, VGPR r -> M=r+8*(L/16).
// ---------------------------------------------------------------------------

typedef __attribute__((ext_vector_type(16))) __bf16 v16bf;
typedef __attribute__((ext_vector_type(8)))  __bf16 v8bf;
typedef __attribute__((ext_vector_type(8)))  float  v8f;
typedef __attribute__((ext_vector_type(4)))  float  v4f;

#define D_MODEL 1024
#define N_HEADS 16
#define DK      64
#define SEQ     2048
#define BATCH   2
#define ROWS    (BATCH * SEQ)   // 4096 token rows

// 1/sqrt(dk) folded with log2(e): softmax computed in exp2 domain so the
// transcendental is a bare v_exp_f32 and the argument is a single v_fma.
#define SCALE_LOG2 0.18033688011112042f   // 0.125 * log2(e)

__device__ __forceinline__ v16bf ld_bf16x16(const __bf16* p0, const __bf16* p1) {
  v8bf lo = *(const v8bf*)p0;
  v8bf hi = *(const v8bf*)p1;
  v16bf r;
#pragma unroll
  for (int i = 0; i < 8; ++i) { r[i] = lo[i]; r[i + 8] = hi[i]; }
  return r;
}

__device__ __forceinline__ v16bf ld_f32x16_cvt(const float* p0, const float* p1) {
  v8f lo = *(const v8f*)p0;
  v8f hi = *(const v8f*)p1;
  v16bf r;
#pragma unroll
  for (int i = 0; i < 8; ++i) { r[i] = (__bf16)lo[i]; r[i + 8] = (__bf16)hi[i]; }
  return r;
}

__device__ __forceinline__ v8f wmma_bf16(v16bf a, v16bf b, v8f c) {
  // (neg_a, A, neg_b, B, c_mod, C, reuse_a, reuse_b)
  return __builtin_amdgcn_wmma_f32_16x16x32_bf16(false, a, false, b, (short)0, c,
                                                 false, false);
}

// ---------------------------------------------------------------------------
// Weight transpose + fp32->bf16:  Wt[n*1024 + k] = (bf16) W[k*1024 + n]
// ---------------------------------------------------------------------------
__global__ void wtrans_kernel(const float* __restrict__ W, __bf16* __restrict__ Wt) {
  int idx = blockIdx.x * 256 + threadIdx.x;      // 1M elements
  int n = idx >> 10;
  int k = idx & 1023;
  Wt[idx] = (__bf16)W[k * D_MODEL + n];
}

// ---------------------------------------------------------------------------
// GEMM: C[M,N] = A[M,K] @ Wt^T + bias.   Each wave computes a 32x64 C tile.
// MODE 0: store bf16 at [B,H,L,DK]      (Q / K for attention)
// MODE 1: store bf16 at [B,H,DK,S]      (V transposed for P@V B-fragments)
// MODE 2: store f32 row-major [M,N]     (final output projection)
// ---------------------------------------------------------------------------
template <int MODE, bool A_IS_BF16>
__global__ void gemm_wmma(const void* __restrict__ Aptr,
                          const __bf16* __restrict__ Bt,
                          const float* __restrict__ bias,
                          void* __restrict__ Dst,
                          int M, int N, int K) {
  const int lane = threadIdx.x & 31;
  const int wid  = blockIdx.x * (blockDim.x >> 5) + (threadIdx.x >> 5);
  const int wavesM = M >> 5;                 // 32 rows per wave
  const int m0 = (wid % wavesM) * 32;
  const int n0 = (wid / wavesM) * 64;
  const int lm = lane & 15;
  const int lh = lane >> 4;

  v8f acc[2][4];
#pragma unroll
  for (int mt = 0; mt < 2; ++mt)
#pragma unroll
    for (int nt = 0; nt < 4; ++nt) acc[mt][nt] = (v8f)0.0f;

  for (int k0 = 0; k0 < K; k0 += 32) {
    v16bf afrag[2];
#pragma unroll
    for (int mt = 0; mt < 2; ++mt) {
      const size_t row = (size_t)(m0 + mt * 16 + lm);
      if (A_IS_BF16) {
        const __bf16* ap = (const __bf16*)Aptr + row * K + k0 + lh * 8;
        afrag[mt] = ld_bf16x16(ap, ap + 16);
      } else {
        const float* ap = (const float*)Aptr + row * K + k0 + lh * 8;
        afrag[mt] = ld_f32x16_cvt(ap, ap + 16);
      }
    }
    v16bf bfrag[4];
#pragma unroll
    for (int nt = 0; nt < 4; ++nt) {
      const size_t col = (size_t)(n0 + nt * 16 + lm);
      const __bf16* bp = Bt + col * K + k0 + lh * 8;
      bfrag[nt] = ld_bf16x16(bp, bp + 16);
    }
#pragma unroll
    for (int mt = 0; mt < 2; ++mt)
#pragma unroll
      for (int nt = 0; nt < 4; ++nt)
        acc[mt][nt] = wmma_bf16(afrag[mt], bfrag[nt], acc[mt][nt]);
  }

  // Epilogue: bias + scatter to the target layout.
#pragma unroll
  for (int mt = 0; mt < 2; ++mt) {
#pragma unroll
    for (int nt = 0; nt < 4; ++nt) {
#pragma unroll
      for (int r = 0; r < 8; ++r) {
        const int m = m0 + mt * 16 + lh * 8 + r;   // token row
        const int n = n0 + nt * 16 + lm;           // feature col
        const float v = acc[mt][nt][r] + bias[n];
        if (MODE == 2) {
          ((float*)Dst)[(size_t)m * N + n] = v;
        } else {
          const int b = m >> 11;           // / SEQ
          const int l = m & (SEQ - 1);
          const int h = n >> 6;            // / DK
          const int d = n & (DK - 1);
          size_t idx;
          if (MODE == 0) idx = (((size_t)b * N_HEADS + h) * SEQ + l) * DK + d;
          else           idx = (((size_t)b * N_HEADS + h) * DK + d) * SEQ + l;
          ((__bf16*)Dst)[idx] = (__bf16)v;
        }
      }
    }
  }
}

// ---------------------------------------------------------------------------
// Per-wave LDS scratch for attention.
//  sl: raw f32 scores, transposed C-layout -> A-layout. Row stride 36 floats
//      keeps the two store lane-halves on disjoint banks (36*8 mod 64 == 32)
//      and row starts 16B-aligned for b128 reads.
//  cl: per-row broadcast channel (corr factor / 1/l).
// ---------------------------------------------------------------------------
struct AttnLds {
  float sl[16][36];
  float cl[16];
};

// One 32-wide S block of flash attention for a 16-row Q tile.
// The raw scores are transposed through LDS once; the entire softmax then
// runs IN-LANE (each lane owns 16 columns of its row, its xor-16 partner the
// other 16), so the only cross-lane traffic is two xor-16 shuffles, and P is
// packed directly into the WMMA A-fragment with no second LDS round-trip.
template <bool MASKED>
__device__ __forceinline__ void attn_block(
    int s0, int l0, int lm, int lh,
    const __bf16* __restrict__ k, const __bf16* __restrict__ vt,
    const v16bf (&aq)[2],
    float& mrow, float& lrow, v8f (&oacc)[4],
    AttnLds& lds) {
  // ---- scores: two 16x16 WMMA tiles, dk=64 contracted in two K=32 steps
  v8f sacc[2];
  sacc[0] = (v8f)0.0f;
  sacc[1] = (v8f)0.0f;
#pragma unroll
  for (int cs = 0; cs < 2; ++cs) {
    const int scol = s0 + cs * 16 + lm;            // key row for this lane
#pragma unroll
    for (int kc = 0; kc < 2; ++kc) {
      const __bf16* kp = k + (size_t)scol * DK + kc * 32 + lh * 8;
      v16bf bk = ld_bf16x16(kp, kp + 16);
      __builtin_prefetch(kp + 32 * DK, 0, 3);      // next S block of K
      sacc[cs] = wmma_bf16(aq[kc], bk, sacc[cs]);
    }
  }

  // ---- transpose raw f32 scores to A-layout through LDS
#pragma unroll
  for (int r = 0; r < 8; ++r) {
    lds.sl[lh * 8 + r][lm]      = sacc[0][r];
    lds.sl[lh * 8 + r][16 + lm] = sacc[1][r];
  }
  __asm__ volatile("s_wait_dscnt 0" ::: "memory");  // wave-local RAW on LDS

  float sv[16];
  {
    v4f c0 = *(const v4f*)&lds.sl[lm][lh * 8];
    v4f c1 = *(const v4f*)&lds.sl[lm][lh * 8 + 4];
    v4f c2 = *(const v4f*)&lds.sl[lm][16 + lh * 8];
    v4f c3 = *(const v4f*)&lds.sl[lm][16 + lh * 8 + 4];
#pragma unroll
    for (int i = 0; i < 4; ++i) {
      sv[i] = c0[i]; sv[4 + i] = c1[i]; sv[8 + i] = c2[i]; sv[12 + i] = c3[i];
    }
  }

  // ---- causal mask (before scale; -inf survives the scale)
  if (MASKED) {
#pragma unroll
    for (int i = 0; i < 16; ++i) {
      const int col = s0 + ((i < 8) ? (lh * 8 + i) : (16 + lh * 8 + i - 8));
      sv[i] = (col > l0 + lm) ? -__builtin_inff() : sv[i];
    }
  }

  // ---- in-lane row max (15 VALU maxes) + one xor-16 shuffle with partner
  float tm[8];
#pragma unroll
  for (int i = 0; i < 8; ++i) tm[i] = fmaxf(sv[i], sv[8 + i]);
#pragma unroll
  for (int st = 4; st >= 1; st >>= 1)
#pragma unroll
    for (int i = 0; i < st; ++i) tm[i] = fmaxf(tm[i], tm[i + st]);
  float bmax = fmaxf(tm[0], __shfl_xor(tm[0], 16, 32));

  // ---- online softmax in exp2 domain: p = exp2(fma(s, c, -m*c))
  const float mnew = fmaxf(mrow, bmax);
  const float corr = exp2f((mrow - mnew) * SCALE_LOG2);
  mrow = mnew;
  const float ms = mnew * SCALE_LOG2;
  float p[16];
  v16bf aP;                                        // P packed straight into A-fragment
#pragma unroll
  for (int i = 0; i < 16; ++i) {
    p[i] = exp2f(fmaf(sv[i], SCALE_LOG2, -ms));
    aP[i] = (__bf16)p[i];
  }
  float ts[8];
#pragma unroll
  for (int i = 0; i < 8; ++i) ts[i] = p[i] + p[8 + i];
#pragma unroll
  for (int st = 4; st >= 1; st >>= 1)
#pragma unroll
    for (int i = 0; i < st; ++i) ts[i] += ts[i + st];
  float rs = ts[0] + __shfl_xor(ts[0], 16, 32);
  lrow = lrow * corr + rs;

  // ---- broadcast corr (per A-row) to the C-layout rows via LDS
  lds.cl[lm] = corr;                               // both halves write same value
  __asm__ volatile("s_wait_dscnt 0" ::: "memory");
  v4f ca = *(const v4f*)&lds.cl[lh * 8];
  v4f cb = *(const v4f*)&lds.cl[lh * 8 + 4];
#pragma unroll
  for (int t = 0; t < 4; ++t)
#pragma unroll
    for (int r = 0; r < 8; ++r)
      oacc[t][r] *= (r < 4) ? ca[r] : cb[r - 4];

  // ---- P @ V : 4 output tiles (d = 0..63), single K=32 contraction step
#pragma unroll
  for (int t = 0; t < 4; ++t) {
    const int d = t * 16 + lm;
    const __bf16* vp = vt + (size_t)d * SEQ + s0 + lh * 8;
    v16bf bv = ld_bf16x16(vp, vp + 16);
    __builtin_prefetch(vp + 32, 0, 3);             // next S block of V^T
    oacc[t] = wmma_bf16(aP, bv, oacc[t]);
  }
}

// ---------------------------------------------------------------------------
// Flash-style causal attention.  One wave = one 16-row Q tile of one (b,h).
// ---------------------------------------------------------------------------
__global__ void attn_kernel(const __bf16* __restrict__ Q,
                            const __bf16* __restrict__ Kmat,
                            const __bf16* __restrict__ Vt,
                            __bf16* __restrict__ O) {
  __shared__ __align__(16) AttnLds lds[4];         // per-wave scratch

  const int lane = threadIdx.x & 31;
  const int w    = threadIdx.x >> 5;
  const int tile = blockIdx.x * 4 + w;             // 4096 tiles total
  const int bh = tile >> 7;                        // SEQ/16 = 128 tiles/head
  const int l0 = (tile & 127) << 4;
  const int lm = lane & 15;
  const int lh = lane >> 4;

  const __bf16* q  = Q    + (size_t)bh * SEQ * DK;
  const __bf16* k  = Kmat + (size_t)bh * SEQ * DK;
  const __bf16* vt = Vt   + (size_t)bh * DK * SEQ;
  AttnLds& L = lds[w];

  // Q A-fragments for the two dk chunks (K = 0..31, 32..63)
  v16bf aq[2];
#pragma unroll
  for (int kc = 0; kc < 2; ++kc) {
    const __bf16* qp = q + (size_t)(l0 + lm) * DK + kc * 32 + lh * 8;
    aq[kc] = ld_bf16x16(qp, qp + 16);
  }

  float mrow = -__builtin_inff();
  float lrow = 0.0f;
  v8f oacc[4];
#pragma unroll
  for (int t = 0; t < 4; ++t) oacc[t] = (v8f)0.0f;

  // Interior blocks (entirely below the diagonal) need no masking:
  // block [s0, s0+31] is full iff s0+31 <= l0.
  const int sFullEnd = (l0 >= 31) ? ((((l0 - 31) >> 5) + 1) << 5) : 0;
  int s0 = 0;
  for (; s0 < sFullEnd; s0 += 32)
    attn_block<false>(s0, l0, lm, lh, k, vt, aq, mrow, lrow, oacc, L);
  for (; s0 < l0 + 16; s0 += 32)
    attn_block<true>(s0, l0, lm, lh, k, vt, aq, mrow, lrow, oacc, L);

  // ---- normalize (1/l broadcast through LDS, same trick) and store bf16
  L.cl[lm] = 1.0f / lrow;
  __asm__ volatile("s_wait_dscnt 0" ::: "memory");
  v4f ia = *(const v4f*)&L.cl[lh * 8];
  v4f ib = *(const v4f*)&L.cl[lh * 8 + 4];

  const int b = bh >> 4;
  const int h = bh & 15;
#pragma unroll
  for (int r = 0; r < 8; ++r) {
    const float inv = (r < 4) ? ia[r] : ib[r - 4];
    const int m = l0 + lh * 8 + r;
#pragma unroll
    for (int t = 0; t < 4; ++t) {
      const int d = t * 16 + lm;
      O[((size_t)(b * SEQ + m)) * D_MODEL + h * DK + d] =
          (__bf16)(oacc[t][r] * inv);
    }
  }
}

// ---------------------------------------------------------------------------
// Host launcher
// ---------------------------------------------------------------------------
extern "C" void kernel_launch(void* const* d_in, const int* in_sizes, int n_in,
                              void* d_out, int out_size, void* d_ws, size_t ws_size,
                              hipStream_t stream) {
  const float* queries = (const float*)d_in[0];
  const float* keys    = (const float*)d_in[1];
  const float* values  = (const float*)d_in[2];
  const float* Wq = (const float*)d_in[3];
  const float* bq = (const float*)d_in[4];
  const float* Wk = (const float*)d_in[5];
  const float* bk = (const float*)d_in[6];
  const float* Wv = (const float*)d_in[7];
  const float* bv = (const float*)d_in[8];
  const float* Wo = (const float*)d_in[9];
  const float* bo = (const float*)d_in[10];
  float* out = (float*)d_out;

  // Workspace layout (bf16 elements): 4 x 4M token buffers + 4 x 1M weights
  __bf16* ws = (__bf16*)d_ws;
  const size_t MT = (size_t)ROWS * D_MODEL;          // 4M elements
  const size_t WT = (size_t)D_MODEL * D_MODEL;       // 1M elements
  __bf16* qb  = ws;
  __bf16* kb  = qb  + MT;
  __bf16* vtb = kb  + MT;
  __bf16* ab  = vtb + MT;
  __bf16* wqt = ab  + MT;
  __bf16* wkt = wqt + WT;
  __bf16* wvt = wkt + WT;
  __bf16* wot = wvt + WT;

  // 1) transpose + convert weights to bf16 [out, in]
  const int wtBlocks = (int)(WT / 256);
  wtrans_kernel<<<wtBlocks, 256, 0, stream>>>(Wq, wqt);
  wtrans_kernel<<<wtBlocks, 256, 0, stream>>>(Wk, wkt);
  wtrans_kernel<<<wtBlocks, 256, 0, stream>>>(Wv, wvt);
  wtrans_kernel<<<wtBlocks, 256, 0, stream>>>(Wo, wot);

  // 2) projections (32x64 per wave, 4 waves per block)
  const int gemmBlocks = (ROWS / 32) * (D_MODEL / 64) / 4;   // 512
  gemm_wmma<0, false><<<gemmBlocks, 128, 0, stream>>>(queries, wqt, bq, qb,
                                                      ROWS, D_MODEL, D_MODEL);
  gemm_wmma<0, false><<<gemmBlocks, 128, 0, stream>>>(keys,    wkt, bk, kb,
                                                      ROWS, D_MODEL, D_MODEL);
  gemm_wmma<1, false><<<gemmBlocks, 128, 0, stream>>>(values,  wvt, bv, vtb,
                                                      ROWS, D_MODEL, D_MODEL);

  // 3) causal flash attention (16-row Q tile per wave, 4 waves per block)
  const int attnBlocks = BATCH * N_HEADS * (SEQ / 16) / 4;   // 1024
  attn_kernel<<<attnBlocks, 128, 0, stream>>>(qb, kb, vtb, ab);

  // 4) output projection -> fp32
  gemm_wmma<2, true><<<gemmBlocks, 128, 0, stream>>>(ab, wot, bo, out,
                                                     ROWS, D_MODEL, D_MODEL);
}
